// ALSDigitalTwin_20959440404679
// MI455X (gfx1250) — matrix operand order repaired and checked by reading, not compiled, not run on hardware
//
#include <hip/hip_runtime.h>
#include <math.h>

// ---------------- problem constants ----------------
#define Bsz 256
#define Ssz 512
#define DTf 64
#define DSf 64
#define Hd  256
#define HEADH 64
#define NEV 5
#define NINT 20
#define NSTATE 6
#define OUTC (NSTATE + NEV * NINT)   // 106

// ---------------- tiling ----------------
#define ROWS 16            // batch rows per workgroup (one WMMA M tile)
#define NBLK (Bsz / ROWS)  // 16 persistent workgroups, no inter-WG sync needed
#define THREADS 512        // 16 wave32s per WGP
#define NT (Hd / 16)       // 16 column tiles -> one per wave
#define KT0 ((DTf + Hd) / 32)  // 10 K-tiles for cell0 (K=320)
#define KT1 ((Hd + Hd) / 32)   // 16 K-tiles for cell1 (K=512)

typedef __bf16 bf16_t;
typedef bf16_t v16bf __attribute__((ext_vector_type(16)));
typedef float  v8f   __attribute__((ext_vector_type(8)));

// explicit global (addrspace 1) pointers so laundered pointers still emit
// global_load_b128 (LOADcnt only) instead of flat_load_b128 (LOADcnt+DScnt)
typedef const unsigned short __attribute__((address_space(1)))* gwptr;
typedef const v16bf __attribute__((address_space(1)))* gfragptr;

__device__ __forceinline__ unsigned short f2bf(float f) {
  unsigned int u = __float_as_uint(f);
  u += 0x7fffu + ((u >> 16) & 1u);   // round-to-nearest-even
  return (unsigned short)(u >> 16);
}

// fast sigmoid: v_exp_f32 + v_rcp_f32, no div-scale/Newton chain
__device__ __forceinline__ float fast_sigmoid(float x) {
  return __builtin_amdgcn_rcpf(1.f + __expf(-x));
}
// fast tanh: tanh(x) = 1 - 2/(exp(2x)+1)
__device__ __forceinline__ float fast_tanh(float x) {
  return 1.f - 2.f * __builtin_amdgcn_rcpf(__expf(2.f * x) + 1.f);
}

struct SL8 { int v[8]; };  // per-lane seq_lens, hoisted out of the t-loop

// =====================================================================
// Kernel 1: pack fp32 weight (dout=256, din) into WMMA-B bf16 fragments.
// Packed index = [kt][nt=16][lane=32][j=16]; element (lane,j) holds
// W[n][k] with k = kt*32 + 8*(lane/16) + (j%8) + 16*(j/8), n = nt*16 + lane%16.
// =====================================================================
__global__ void pack_weight_kernel(const float* __restrict__ W,
                                   unsigned short* __restrict__ P, int din) {
  int total = (din / 32) * NT * 512;
  for (int idx = blockIdx.x * blockDim.x + threadIdx.x; idx < total;
       idx += gridDim.x * blockDim.x) {
    int kt   = idx >> 13;        // / (16*512)
    int rem  = idx & 8191;
    int nt   = rem >> 9;
    int rem2 = rem & 511;
    int lane = rem2 >> 4;
    int j    = rem2 & 15;
    int k = kt * 32 + ((lane >> 4) << 3) + (j & 7) + ((j >> 3) << 4);
    int n = nt * 16 + (lane & 15);
    P[idx] = f2bf(W[n * din + k]);
  }
}

// =====================================================================
// Build packed A = [x_imp | gamma_h * h]  (bf16, WMMA A-fragment layout).
// A layout (16-bit 16x32): lane = m + 16*g, element j ->
//   k = kt*32 + 8*g + (j%8) + 16*(j/8).
// Also scales hT in place: hT becomes hs = gamma_h * h (raw h dead after).
// =====================================================================
template <int DIN, int KT, bool CELL0>
__device__ __forceinline__ void build_a(
    float (*hT)[Hd], const float (*xT)[Hd], unsigned short* aP, int b0, int t,
    const float* __restrict__ temporal, const float* __restrict__ maskp,
    const float* __restrict__ tdelta,
    const float* __restrict__ wgh_w, const float* __restrict__ wgh_b,
    const float* __restrict__ wgx_w, const float* __restrict__ wgx_b,
    const float* __restrict__ xmean) {
  for (int idx = threadIdx.x; idx < KT * 512; idx += THREADS) {
    int kt    = idx >> 9;
    int rem   = idx & 511;
    int laneD = rem >> 4;
    int j     = rem & 15;
    int g = laneD >> 4;
    int m = laneD & 15;
    int k = kt * 32 + (g << 3) + (j & 7) + ((j >> 3) << 4);
    int bb = b0 + m;
    float d = tdelta[bb * Ssz + t];
    float v;
    if (k < DIN) {
      if constexpr (CELL0) {
        float x  = temporal[(bb * Ssz + t) * DTf + k];
        float mk = maskp[(bb * Ssz + t) * DTf + k];
        float gx = __expf(-fmaxf(0.f, d * wgx_w[k] + wgx_b[k]));
        v = mk * x + (1.f - mk) * (gx * x + (1.f - gx) * xmean[k]);
        if (t + 1 < Ssz) {  // overlap next-step HBM/L2 traffic with compute
          __builtin_prefetch(&temporal[(bb * Ssz + t + 1) * DTf + k], 0, 1);
          __builtin_prefetch(&maskp[(bb * Ssz + t + 1) * DTf + k], 0, 1);
        }
      } else {
        v = xT[m][k];  // cell1 input = h0 (mask==1 -> x_imp == x)
      }
    } else {
      int j2 = k - DIN;
      float gh = __expf(-fmaxf(0.f, d * wgh_w[j2] + wgh_b[j2]));
      float hs = gh * hT[m][j2];
      hT[m][j2] = hs;  // in-place: hT now holds scaled state
      v = hs;
    }
    aP[idx] = f2bf(v);
  }
}

// =====================================================================
// One GRU-D cell: z/r GEMMs -> sigmoid -> scatter r*hs into A -> h~ GEMM
// -> tanh -> state update. All epilogues operate directly on the WMMA
// accumulator fragment: VGPR e, lane -> (m = e + 8*(lane/16), n = w*16+lane%16)
// =====================================================================
template <int KT, int DIN>
__device__ __forceinline__ void gemm_update(
    float (*hT)[Hd], unsigned short* aP,
    gwptr pz, gwptr pr, gwptr ph,
    float zbias, float rbias, float hbias,
    float* __restrict__ finalDst, int writeFinal, SL8 sl, int b0, int t) {
  int tid  = threadIdx.x;
  int w    = tid >> 5;     // wave id = column tile
  int lane = tid & 31;
  int nloc = lane & 15;
  int g    = lane >> 4;
  int n    = w * 16 + nloc;

  v8f az = {}, ar = {};
#pragma unroll 2
  for (int kt = 0; kt < KT; ++kt) {
    v16bf a  = *(const v16bf*)&aP[kt * 512 + lane * 16];
    v16bf wz = *(gfragptr)&pz[(kt * NT + w) * 512 + lane * 16];
    v16bf wr = *(gfragptr)&pr[(kt * NT + w) * 512 + lane * 16];
    az = __builtin_amdgcn_wmma_f32_16x16x32_bf16(false, a, false, wz, (short)0,
                                                 az, false, false);
    ar = __builtin_amdgcn_wmma_f32_16x16x32_bf16(false, a, false, wr, (short)0,
                                                 ar, false, false);
  }
  __syncthreads();  // all waves done reading aP before we overwrite its h-part

  float zv[8];
  // destination slot in packed A for column k = DIN + n
  int k   = DIN + n;
  int ktD = k >> 5, kk = k & 31;
  int gD  = (kk >> 3) & 1;
  int jD  = (kk & 7) + ((kk >> 4) << 3);
#pragma unroll
  for (int e = 0; e < 8; ++e) {
    int m = e + (g << 3);
    float z = fast_sigmoid(az[e] + zbias);
    float r = fast_sigmoid(ar[e] + rbias);
    zv[e] = z;
    float rh = r * hT[m][n];  // hT holds hs
    aP[ktD * 512 + (m + (gD << 4)) * 16 + jD] = f2bf(rh);
  }
  __syncthreads();

  v8f ah = {};
#pragma unroll 2
  for (int kt = 0; kt < KT; ++kt) {
    v16bf a  = *(const v16bf*)&aP[kt * 512 + lane * 16];
    v16bf wh = *(gfragptr)&ph[(kt * NT + w) * 512 + lane * 16];
    ah = __builtin_amdgcn_wmma_f32_16x16x32_bf16(false, a, false, wh, (short)0,
                                                 ah, false, false);
  }
#pragma unroll
  for (int e = 0; e < 8; ++e) {
    int m = e + (g << 3);
    float htl = fast_tanh(ah[e] + hbias);
    float hn  = (1.f - zv[e]) * hT[m][n] + zv[e] * htl;
    hT[m][n]  = hn;
    if (writeFinal && t == sl.v[e] - 1)
      finalDst[(b0 + m) * (2 * Hd) + Hd + n] = hn;  // second half of concat buf
  }
  __syncthreads();
}

// =====================================================================
// Kernel 2: persistent recurrent kernel. 16 WGs x 16 batch rows, all 512
// steps. State lives in LDS; weights stream from L2 (bf16 packed).
// =====================================================================
__global__ __launch_bounds__(THREADS) void grud_recurrent_kernel(
    const float* __restrict__ temporal, const float* __restrict__ maskp,
    const float* __restrict__ tdelta, const int* __restrict__ seqLens,
    const float* __restrict__ bz0, const float* __restrict__ br0,
    const float* __restrict__ bh0, const float* __restrict__ wgh0w,
    const float* __restrict__ wgh0b, const float* __restrict__ wgx0w,
    const float* __restrict__ wgx0b, const float* __restrict__ xmean0,
    const float* __restrict__ bz1, const float* __restrict__ br1,
    const float* __restrict__ bh1, const float* __restrict__ wgh1w,
    const float* __restrict__ wgh1b,
    const unsigned short* pz0_, const unsigned short* pr0_,
    const unsigned short* ph0_, const unsigned short* pz1_,
    const unsigned short* pr1_, const unsigned short* ph1_,
    float* __restrict__ concatBuf) {
  __shared__ float h0T[ROWS][Hd];
  __shared__ float h1T[ROWS][Hd];
  __shared__ unsigned short aPack[KT1 * 512];  // 16 KB, reused by both cells

  int b0 = blockIdx.x * ROWS;
  int tid = threadIdx.x;
  int lane = tid & 31, g = tid >> 4 & 1;
  g = (tid & 31) >> 4;
  int n_me = (tid >> 5) * 16 + (tid & 15);

  // hoist all loop-invariant per-lane scalars into registers BEFORE the
  // t-loop (the per-step memory clobber would otherwise force re-loads)
  float c0zb = bz0[n_me], c0rb = br0[n_me], c0hb = bh0[n_me];
  float c1zb = bz1[n_me], c1rb = br1[n_me], c1hb = bh1[n_me];
  SL8 sl;
#pragma unroll
  for (int e = 0; e < 8; ++e) sl.v[e] = seqLens[b0 + e + (g << 3)];

  gwptr pz0 = (gwptr)pz0_, pr0 = (gwptr)pr0_, ph0 = (gwptr)ph0_;
  gwptr pz1 = (gwptr)pz1_, pr1 = (gwptr)pr1_, ph1 = (gwptr)ph1_;

  for (int i = tid; i < ROWS * Hd; i += THREADS) {
    ((float*)h0T)[i] = 0.f;
    ((float*)h1T)[i] = 0.f;
  }
  __syncthreads();

  for (int t = 0; t < Ssz; ++t) {
    // Opaque barrier: makes the packed-weight pointers loop-variant and the
    // asm a may-write, so LICM cannot hoist 600+ VGPRs of weight fragments
    // out of the 512-step loop (which caused scratch spills). Weights are
    // deliberately re-streamed from the 192MB L2 every step.
    asm volatile("" : "+s"(pz0), "+s"(pr0), "+s"(ph0),
                      "+s"(pz1), "+s"(pr1), "+s"(ph1) :: "memory");

    build_a<DTf, KT0, true>(h0T, nullptr, aPack, b0, t, temporal, maskp, tdelta,
                            wgh0w, wgh0b, wgx0w, wgx0b, xmean0);
    __syncthreads();
    gemm_update<KT0, DTf>(h0T, aPack, pz0, pr0, ph0, c0zb, c0rb, c0hb, nullptr,
                          0, sl, b0, t);
    build_a<Hd, KT1, false>(h1T, h0T, aPack, b0, t, temporal, maskp, tdelta,
                            wgh1w, wgh1b, nullptr, nullptr, nullptr);
    __syncthreads();
    gemm_update<KT1, Hd>(h1T, aPack, pz1, pr1, ph1, c1zb, c1rb, c1hb, concatBuf,
                         1, sl, b0, t);
  }
}

// =====================================================================
// Kernel 3: generic linear (+optional relu). Head work is ~0.2 GFLOP total.
// =====================================================================
__global__ void linear_kernel(const float* __restrict__ X, int ldx,
                              const float* __restrict__ W,
                              const float* __restrict__ bias,
                              float* __restrict__ Y, int ldy, int rows, int K,
                              int N, int doRelu) {
  int idx = blockIdx.x * blockDim.x + threadIdx.x;
  if (idx >= rows * N) return;
  int r = idx / N, n = idx % N;
  const float* x  = X + r * ldx;
  const float* wr = W + n * K;
  float acc = bias[n];
  for (int kk = 0; kk < K; ++kk) acc = fmaf(x[kk], wr[kk], acc);
  Y[r * ldy + n] = doRelu ? fmaxf(acc, 0.f) : acc;
}

// ---------------- input index map (setup_inputs insertion order) -------
enum {
  IN_STATIC = 0, IN_TEMPORAL, IN_MASK, IN_TDELTA, IN_SEQLEN,
  C0_WZ_W, C0_WZ_B, C0_WR_W, C0_WR_B, C0_WH_W, C0_WH_B,
  C0_WGH_W, C0_WGH_B, C0_WGX_W, C0_WGX_B, C0_XMEAN,
  C1_WZ_W, C1_WZ_B, C1_WR_W, C1_WR_B, C1_WH_W, C1_WH_B,
  C1_WGH_W, C1_WGH_B, C1_WGX_W, C1_WGX_B, C1_XMEAN,
  S1_W, S1_B, S2_W, S2_B, FU_W, FU_B,
  ST0_W, ST0_B, ST1_W, ST1_B, ST2_W, ST2_B,
  SURV_BASE  // surv[e][j]: w = SURV_BASE + e*6 + j*2, b = +1
};

extern "C" void kernel_launch(void* const* d_in, const int* in_sizes, int n_in,
                              void* d_out, int out_size, void* d_ws,
                              size_t ws_size, hipStream_t stream) {
  (void)in_sizes; (void)n_in; (void)out_size; (void)ws_size;
  auto fin = [&](int i) { return (const float*)d_in[i]; };

  // ---- workspace layout (bytes) ----
  char* ws = (char*)d_ws;
  const size_t SZ0 = (size_t)KT0 * NT * 512 * 2;  // 163840 B per cell0 matrix
  const size_t SZ1 = (size_t)KT1 * NT * 512 * 2;  // 262144 B per cell1 matrix
  unsigned short* pz0 = (unsigned short*)(ws + 0);
  unsigned short* pr0 = (unsigned short*)(ws + SZ0);
  unsigned short* ph0 = (unsigned short*)(ws + 2 * SZ0);
  unsigned short* pz1 = (unsigned short*)(ws + 3 * SZ0);
  unsigned short* pr1 = (unsigned short*)(ws + 3 * SZ0 + SZ1);
  unsigned short* ph1 = (unsigned short*)(ws + 3 * SZ0 + 2 * SZ1);
  size_t off = 3 * SZ0 + 3 * SZ1;
  float* concatBuf = (float*)(ws + off); off += (size_t)Bsz * 2 * Hd * 4;  // [256][512]
  float* s1buf     = (float*)(ws + off); off += (size_t)Bsz * Hd * 4;
  float* latent    = (float*)(ws + off); off += (size_t)Bsz * Hd * 4;
  float* hA        = (float*)(ws + off); off += (size_t)Bsz * HEADH * 4;
  float* hB        = (float*)(ws + off); off += (size_t)Bsz * HEADH * 4;

  // ---- 1) pack recurrent weights to bf16 WMMA-B fragments ----
  {
    int t0 = KT0 * NT * 512, t1 = KT1 * NT * 512;
    int g0 = (t0 + 511) / 512, g1 = (t1 + 511) / 512;
    pack_weight_kernel<<<g0, 512, 0, stream>>>(fin(C0_WZ_W), pz0, DTf + Hd);
    pack_weight_kernel<<<g0, 512, 0, stream>>>(fin(C0_WR_W), pr0, DTf + Hd);
    pack_weight_kernel<<<g0, 512, 0, stream>>>(fin(C0_WH_W), ph0, DTf + Hd);
    pack_weight_kernel<<<g1, 512, 0, stream>>>(fin(C1_WZ_W), pz1, Hd + Hd);
    pack_weight_kernel<<<g1, 512, 0, stream>>>(fin(C1_WR_W), pr1, Hd + Hd);
    pack_weight_kernel<<<g1, 512, 0, stream>>>(fin(C1_WH_W), ph1, Hd + Hd);
  }

  // ---- 2) persistent recurrent kernel ----
  grud_recurrent_kernel<<<NBLK, THREADS, 0, stream>>>(
      fin(IN_TEMPORAL), fin(IN_MASK), fin(IN_TDELTA), (const int*)d_in[IN_SEQLEN],
      fin(C0_WZ_B), fin(C0_WR_B), fin(C0_WH_B), fin(C0_WGH_W), fin(C0_WGH_B),
      fin(C0_WGX_W), fin(C0_WGX_B), fin(C0_XMEAN),
      fin(C1_WZ_B), fin(C1_WR_B), fin(C1_WH_B), fin(C1_WGH_W), fin(C1_WGH_B),
      pz0, pr0, ph0, pz1, pr1, ph1, concatBuf);

  // ---- 3) heads ----
  auto lin = [&](const float* X, int ldx, const float* W, const float* b,
                 float* Y, int ldy, int rows, int K, int N, int relu) {
    int tot = rows * N;
    linear_kernel<<<(tot + 255) / 256, 256, 0, stream>>>(X, ldx, W, b, Y, ldy,
                                                         rows, K, N, relu);
  };
  float* out = (float*)d_out;  // [256][106]
  // static path: s1 = relu(static W1), s2 = relu(s1 W2) -> concat[:, :256]
  lin(fin(IN_STATIC), DSf, fin(S1_W), fin(S1_B), s1buf, Hd, Bsz, DSf, Hd, 1);
  lin(s1buf, Hd, fin(S2_W), fin(S2_B), concatBuf, 2 * Hd, Bsz, Hd, Hd, 1);
  // latent = relu(concat Wf)
  lin(concatBuf, 2 * Hd, fin(FU_W), fin(FU_B), latent, Hd, Bsz, 2 * Hd, Hd, 1);
  // state head -> out cols [0,6)
  lin(latent, Hd, fin(ST0_W), fin(ST0_B), hA, HEADH, Bsz, Hd, HEADH, 1);
  lin(hA, HEADH, fin(ST1_W), fin(ST1_B), hB, HEADH, Bsz, HEADH, HEADH, 1);
  lin(hB, HEADH, fin(ST2_W), fin(ST2_B), out + 0, OUTC, Bsz, HEADH, NSTATE, 0);
  // hazard heads -> out cols [6 + 20e, ...)
  for (int e = 0; e < NEV; ++e) {
    int wbase = SURV_BASE + e * 6;
    lin(latent, Hd, fin(wbase + 0), fin(wbase + 1), hA, HEADH, Bsz, Hd, HEADH, 1);
    lin(hA, HEADH, fin(wbase + 2), fin(wbase + 3), hB, HEADH, Bsz, HEADH, HEADH, 1);
    lin(hB, HEADH, fin(wbase + 4), fin(wbase + 5), out + NSTATE + NINT * e, OUTC,
        Bsz, HEADH, NINT, 0);
  }
}
